// CLFMv2_NoSmoothnessLoss_39436389712130
// MI455X (gfx1250) — matrix operand order repaired and checked by reading, not compiled, not run on hardware
//
#include <hip/hip_runtime.h>
#include <hip/hip_bf16.h>

// ---------------------------------------------------------------------------
// Sizes from the reference
// ---------------------------------------------------------------------------
#define B_   64
#define L_   12
#define N_   4096
#define C_   3
#define FD_  32
#define HD_  64
#define OUT_ 12
#define TID_ 288
#define DIW_ 7
#define TD_  16
#define STEPS_ 4
#define COLS_ (B_ * FD_)     // 2048 : folded (batch, feature) columns

typedef __attribute__((ext_vector_type(8)))  _Float16 v8h;
typedef __attribute__((ext_vector_type(16))) _Float16 v16h;
typedef __attribute__((ext_vector_type(8)))  float    v8f;

// ---------------------------------------------------------------------------
// 1) Adjacency: A[n][m] = softmax_m( relu(<e_n, e_m>) )   (stored f16)
//    one block (256 threads) per row n
// ---------------------------------------------------------------------------
__global__ __launch_bounds__(256) void adjacency_kernel(
    const float* __restrict__ emb, _Float16* __restrict__ A) {
  __shared__ float sc[N_];
  __shared__ float red[256];
  __shared__ float en[10];
  const int n = blockIdx.x;
  const int t = threadIdx.x;
  if (t < 10) en[t] = emb[n * 10 + t];
  __syncthreads();

  float lmax = 0.0f;   // relu() >= 0
  for (int m = t; m < N_; m += 256) {
    const float* em = emb + m * 10;
    float s = 0.0f;
#pragma unroll
    for (int e = 0; e < 10; ++e) s += en[e] * em[e];
    s = fmaxf(s, 0.0f);
    sc[m] = s;
    lmax = fmaxf(lmax, s);
  }
  red[t] = lmax;
  __syncthreads();
  for (int off = 128; off > 0; off >>= 1) {
    if (t < off) red[t] = fmaxf(red[t], red[t + off]);
    __syncthreads();
  }
  const float mx = red[0];
  __syncthreads();

  float lsum = 0.0f;
  for (int m = t; m < N_; m += 256) {
    float e = __expf(sc[m] - mx);
    sc[m] = e;
    lsum += e;
  }
  red[t] = lsum;
  __syncthreads();
  for (int off = 128; off > 0; off >>= 1) {
    if (t < off) red[t] += red[t + off];
    __syncthreads();
  }
  const float inv = 1.0f / red[0];
  __syncthreads();

  for (int m = t; m < N_; m += 256)
    A[(size_t)n * N_ + m] = (_Float16)(sc[m] * inv);
}

// ---------------------------------------------------------------------------
// 2) Encoder: field = relu(x @ enc_w1 + b1) @ enc_w2 + b2 + temp @ t2f_w + t2f_b
//    field stored as [c = b*32+f][m] (f32 + f16 mirror), state zeroed.
// ---------------------------------------------------------------------------
__global__ __launch_bounds__(256) void encode_kernel(
    const float* __restrict__ hist,
    const float* __restrict__ tid_emb, const float* __restrict__ diw_emb,
    const float* __restrict__ t2f_w, const float* __restrict__ t2f_b,
    const float* __restrict__ w1, const float* __restrict__ b1,
    const float* __restrict__ w2, const float* __restrict__ b2,
    float* __restrict__ field, _Float16* __restrict__ FT,
    float* __restrict__ state) {
  __shared__ float sW1[36 * 64];
  __shared__ float sW2[64 * 32];
  __shared__ float sT2F[32 * 32];
  __shared__ float sB1[64];
  __shared__ float sB2[32];
  __shared__ float sTB[32];
  const int t = threadIdx.x;
  for (int i = t; i < 36 * 64; i += 256) sW1[i] = w1[i];
  for (int i = t; i < 64 * 32; i += 256) sW2[i] = w2[i];
  for (int i = t; i < 32 * 32; i += 256) sT2F[i] = t2f_w[i];
  if (t < 64) sB1[t] = b1[t];
  if (t < 32) { sB2[t] = b2[t]; sTB[t] = t2f_b[t]; }
  __syncthreads();

  const int tkn = blockIdx.x * 256 + t;
  const int b = tkn >> 12;       // / 4096
  const int m = tkn & (N_ - 1);  // % 4096

  float x[36];
#pragma unroll
  for (int l = 0; l < L_; ++l) {
    const size_t base = ((size_t)(b * L_ + l) * N_ + m) * C_;
#pragma unroll
    for (int c = 0; c < C_; ++c) x[l * 3 + c] = hist[base + c];
  }

  float acc[FD_];
#pragma unroll
  for (int f = 0; f < FD_; ++f) acc[f] = sB2[f];

  for (int j = 0; j < HD_; ++j) {
    float h = sB1[j];
#pragma unroll
    for (int q = 0; q < 36; ++q) h += x[q] * sW1[q * 64 + j];
    h = fmaxf(h, 0.0f);
#pragma unroll
    for (int f = 0; f < FD_; ++f) acc[f] += h * sW2[j * 32 + f];
  }

  // temporal embeddings (from last history step, channels 1 and 2)
  const size_t lastb = ((size_t)(b * L_ + (L_ - 1)) * N_ + m) * C_;
  int ti = (int)(hist[lastb + 1] * (float)TID_);
  int di = (int)(hist[lastb + 2] * (float)DIW_);
  ti = ti < 0 ? 0 : (ti > TID_ - 1 ? TID_ - 1 : ti);
  di = di < 0 ? 0 : (di > DIW_ - 1 ? DIW_ - 1 : di);
#pragma unroll
  for (int f = 0; f < FD_; ++f) acc[f] += sTB[f];
  for (int q = 0; q < TD_; ++q) {
    const float tq = tid_emb[ti * TD_ + q];
#pragma unroll
    for (int f = 0; f < FD_; ++f) acc[f] += tq * sT2F[q * 32 + f];
  }
  for (int q = 0; q < TD_; ++q) {
    const float dq = diw_emb[di * TD_ + q];
#pragma unroll
    for (int f = 0; f < FD_; ++f) acc[f] += dq * sT2F[(TD_ + q) * 32 + f];
  }

#pragma unroll
  for (int f = 0; f < FD_; ++f) {
    const size_t idx = (size_t)(b * FD_ + f) * N_ + m;
    field[idx] = acc[f];
    FT[idx] = (_Float16)acc[f];
    state[idx] = 0.0f;
  }
}

// ---------------------------------------------------------------------------
// 3) WMMA GEMM:  LF[n][c] = sum_m A[n][m] * FT[c][m]      (f16 x f16 -> f32)
//    block = 8 waves, 128x128 tile; wave: 2(M) x 4(N) 16x16 subtiles.
//    Software-pipelined: fragments for iteration k+32 are loaded BEFORE the
//    WMMA block of iteration k, so every load has a full 8-WMMA window of
//    latency hiding (double-buffered fragment registers, ~180 VGPRs).
// ---------------------------------------------------------------------------
__device__ __forceinline__ void load_frags(
    const _Float16* const* aRow, const _Float16* const* bRow,
    int k0, int aSel, int bSel, v16h* aF, v16h* bF) {
#pragma unroll
  for (int i = 0; i < 2; ++i) {
    union { v16h v; v8h h[2]; } u;
    u.h[0] = *(const v8h*)(aRow[i] + k0 + aSel);
    u.h[1] = *(const v8h*)(aRow[i] + k0 + aSel + 16);
    aF[i] = u.v;
  }
#pragma unroll
  for (int j = 0; j < 4; ++j) {
    union { v16h v; v8h h[2]; } u;
    u.h[0] = *(const v8h*)(bRow[j] + k0 + bSel);
    u.h[1] = *(const v8h*)(bRow[j] + k0 + bSel + 16);
    bF[j] = u.v;
  }
}

__global__ __launch_bounds__(256) void gemm_lf_kernel(
    const _Float16* __restrict__ A, const _Float16* __restrict__ FT,
    float* __restrict__ Out) {
  const int lane = threadIdx.x & 31;
  const int w = threadIdx.x >> 5;
  const int waveM = w & 3;    // 4 waves along M
  const int waveN = w >> 2;   // 2 waves along N
  const int rowBase = blockIdx.x * 128 + waveM * 32;
  const int colBase = blockIdx.y * 128 + waveN * 64;
  const int lr = lane & 15;
  const int hi = lane >> 4;
  const int aSel = hi * 8;    // K sub-offset for A fragment
  const int bSel = hi * 8;    // K sub-offset for B fragment

  v8f acc[2][4];
#pragma unroll
  for (int i = 0; i < 2; ++i)
#pragma unroll
    for (int j = 0; j < 4; ++j)
#pragma unroll
      for (int r = 0; r < 8; ++r) acc[i][j][r] = 0.0f;

  const _Float16* aRow[2];
  aRow[0] = A + (size_t)(rowBase + lr) * N_;
  aRow[1] = A + (size_t)(rowBase + 16 + lr) * N_;
  const _Float16* bRow[4];
#pragma unroll
  for (int j = 0; j < 4; ++j)
    bRow[j] = FT + (size_t)(colBase + j * 16 + lr) * N_;

  v16h aF[2], bF[4];
  load_frags(aRow, bRow, 0, aSel, bSel, aF, bF);   // prologue

#pragma unroll 2
  for (int k0 = 32; k0 < N_; k0 += 32) {
    v16h aN[2], bN[4];
    load_frags(aRow, bRow, k0, aSel, bSel, aN, bN);  // prefetch next iter
#pragma unroll
    for (int j = 0; j < 4; ++j)
#pragma unroll
      for (int i = 0; i < 2; ++i)
        acc[i][j] = __builtin_amdgcn_wmma_f32_16x16x32_f16(
            false, aF[i], false, bF[j], (short)0, acc[i][j], false, false);
#pragma unroll
    for (int i = 0; i < 2; ++i) aF[i] = aN[i];
#pragma unroll
    for (int j = 0; j < 4; ++j) bF[j] = bN[j];
  }
  // epilogue: last iteration's WMMAs
#pragma unroll
  for (int j = 0; j < 4; ++j)
#pragma unroll
    for (int i = 0; i < 2; ++i)
      acc[i][j] = __builtin_amdgcn_wmma_f32_16x16x32_f16(
          false, aF[i], false, bF[j], (short)0, acc[i][j], false, false);

  // D layout: VGPR r holds M=r (lanes 0-15) / M=r+8 (lanes 16-31), N=lane&15
#pragma unroll
  for (int i = 0; i < 2; ++i)
#pragma unroll
    for (int j = 0; j < 4; ++j) {
      const int n0 = rowBase + i * 16 + hi * 8;
      const int c = colBase + j * 16 + lr;
#pragma unroll
      for (int r = 0; r < 8; ++r)
        Out[(size_t)(n0 + r) * COLS_ + c] = acc[i][j][r];
    }
}

// ---------------------------------------------------------------------------
// 4) PDE + state-space update (one thread per token):
//    L_F = LF_raw - field ; N_F = relu(field@pw1+pb1)@pw2+pb2
//    fe  = field + (a*L_F + (1-a)*N_F)*dt
//    st' = tanh(fe@win + st@wst + sb) ; field' = fe + st'@wout + sbout
// ---------------------------------------------------------------------------
__global__ __launch_bounds__(256) void pde_step_kernel(
    const float* __restrict__ LF, const float* __restrict__ pde_mix,
    const float* __restrict__ pw1, const float* __restrict__ pb1,
    const float* __restrict__ pw2, const float* __restrict__ pb2,
    const float* __restrict__ win, const float* __restrict__ wst,
    const float* __restrict__ sb, const float* __restrict__ wout,
    const float* __restrict__ sbout,
    float* __restrict__ field, _Float16* __restrict__ FT,
    float* __restrict__ state) {
  __shared__ float sPW1[32 * 64];
  __shared__ float sPW2[64 * 32];
  __shared__ float sWin[32 * 32];
  __shared__ float sWst[32 * 32];
  __shared__ float sWout[32 * 32];
  __shared__ float sPB1[64];
  __shared__ float sPB2[32];
  __shared__ float sSB[32];
  __shared__ float sSBO[32];
  const int t = threadIdx.x;
  for (int i = t; i < 32 * 64; i += 256) sPW1[i] = pw1[i];
  for (int i = t; i < 64 * 32; i += 256) sPW2[i] = pw2[i];
  for (int i = t; i < 32 * 32; i += 256) {
    sWin[i] = win[i]; sWst[i] = wst[i]; sWout[i] = wout[i];
  }
  if (t < 64) sPB1[t] = pb1[t];
  if (t < 32) { sPB2[t] = pb2[t]; sSB[t] = sb[t]; sSBO[t] = sbout[t]; }
  __syncthreads();

  const int tkn = blockIdx.x * 256 + t;
  const int b = tkn >> 12;
  const int m = tkn & (N_ - 1);
  const float alpha = 1.0f / (1.0f + __expf(-pde_mix[0]));
  const float dt = 1.0f / (float)STEPS_;

  float fld[FD_];
#pragma unroll
  for (int f = 0; f < FD_; ++f)
    fld[f] = field[(size_t)(b * FD_ + f) * N_ + m];

  // N_F
  float nf[FD_];
#pragma unroll
  for (int f = 0; f < FD_; ++f) nf[f] = sPB2[f];
  for (int j = 0; j < HD_; ++j) {
    float h = sPB1[j];
#pragma unroll
    for (int f = 0; f < FD_; ++f) h += fld[f] * sPW1[f * 64 + j];
    h = fmaxf(h, 0.0f);
#pragma unroll
    for (int f = 0; f < FD_; ++f) nf[f] += h * sPW2[j * 32 + f];
  }

  float fe[FD_];
#pragma unroll
  for (int f = 0; f < FD_; ++f) {
    const float lf = LF[(size_t)m * COLS_ + b * FD_ + f] - fld[f];
    fe[f] = fld[f] + (alpha * lf + (1.0f - alpha) * nf[f]) * dt;
  }

  // state update
  float accst[FD_];
#pragma unroll
  for (int f2 = 0; f2 < FD_; ++f2) accst[f2] = sSB[f2];
  for (int f = 0; f < FD_; ++f) {
    const float st = state[(size_t)(b * FD_ + f) * N_ + m];
    const float fef = fe[f];
#pragma unroll
    for (int f2 = 0; f2 < FD_; ++f2)
      accst[f2] += fef * sWin[f * 32 + f2] + st * sWst[f * 32 + f2];
  }
#pragma unroll
  for (int f2 = 0; f2 < FD_; ++f2) accst[f2] = tanhf(accst[f2]);

  float outv[FD_];
#pragma unroll
  for (int f = 0; f < FD_; ++f) outv[f] = fe[f] + sSBO[f];
  for (int f2 = 0; f2 < FD_; ++f2) {
    const float s2 = accst[f2];
#pragma unroll
    for (int f = 0; f < FD_; ++f) outv[f] += s2 * sWout[f2 * 32 + f];
  }

#pragma unroll
  for (int f = 0; f < FD_; ++f) {
    const size_t idx = (size_t)(b * FD_ + f) * N_ + m;
    field[idx] = outv[f];
    FT[idx] = (_Float16)outv[f];
    state[idx] = accst[f];
  }
}

// ---------------------------------------------------------------------------
// 5) Decoder: pred = relu(field@dw1+db1)@dw2+db2 ; out[b][o][n] = pred[b][n][o]
// ---------------------------------------------------------------------------
__global__ __launch_bounds__(256) void decode_kernel(
    const float* __restrict__ field,
    const float* __restrict__ w1, const float* __restrict__ b1,
    const float* __restrict__ w2, const float* __restrict__ b2,
    float* __restrict__ out) {
  __shared__ float sW1[32 * 64];
  __shared__ float sW2[64 * OUT_];
  __shared__ float sB1[64];
  __shared__ float sB2[OUT_];
  const int t = threadIdx.x;
  for (int i = t; i < 32 * 64; i += 256) sW1[i] = w1[i];
  for (int i = t; i < 64 * OUT_; i += 256) sW2[i] = w2[i];
  if (t < 64) sB1[t] = b1[t];
  if (t < OUT_) sB2[t] = b2[t];
  __syncthreads();

  const int tkn = blockIdx.x * 256 + t;
  const int b = tkn >> 12;
  const int m = tkn & (N_ - 1);

  float fld[FD_];
#pragma unroll
  for (int f = 0; f < FD_; ++f)
    fld[f] = field[(size_t)(b * FD_ + f) * N_ + m];

  float pr[OUT_];
#pragma unroll
  for (int o = 0; o < OUT_; ++o) pr[o] = sB2[o];
  for (int j = 0; j < HD_; ++j) {
    float h = sB1[j];
#pragma unroll
    for (int f = 0; f < FD_; ++f) h += fld[f] * sW1[f * 64 + j];
    h = fmaxf(h, 0.0f);
#pragma unroll
    for (int o = 0; o < OUT_; ++o) pr[o] += h * sW2[j * OUT_ + o];
  }
#pragma unroll
  for (int o = 0; o < OUT_; ++o)
    out[((size_t)b * OUT_ + o) * N_ + m] = pr[o];
}

// ---------------------------------------------------------------------------
// Launch
// ---------------------------------------------------------------------------
extern "C" void kernel_launch(void* const* d_in, const int* in_sizes, int n_in,
                              void* d_out, int out_size, void* d_ws, size_t ws_size,
                              hipStream_t stream) {
  const float* hist     = (const float*)d_in[0];
  const float* tid_emb  = (const float*)d_in[5];
  const float* diw_emb  = (const float*)d_in[6];
  const float* t2f_w    = (const float*)d_in[7];
  const float* t2f_b    = (const float*)d_in[8];
  const float* enc_w1   = (const float*)d_in[9];
  const float* enc_b1   = (const float*)d_in[10];
  const float* enc_w2   = (const float*)d_in[11];
  const float* enc_b2   = (const float*)d_in[12];
  const float* node_emb = (const float*)d_in[13];
  const float* pde_w1   = (const float*)d_in[14];
  const float* pde_b1   = (const float*)d_in[15];
  const float* pde_w2   = (const float*)d_in[16];
  const float* pde_b2   = (const float*)d_in[17];
  const float* ss_win   = (const float*)d_in[18];
  const float* ss_wst   = (const float*)d_in[19];
  const float* ss_b     = (const float*)d_in[20];
  const float* ss_wout  = (const float*)d_in[21];
  const float* ss_bout  = (const float*)d_in[22];
  const float* dec_w1   = (const float*)d_in[23];
  const float* dec_b1   = (const float*)d_in[24];
  const float* dec_w2   = (const float*)d_in[25];
  const float* dec_b2   = (const float*)d_in[26];
  const float* pde_mix  = (const float*)d_in[27];

  // Workspace layout (151 MB total)
  char* ws = (char*)d_ws;
  _Float16* A     = (_Float16*)(ws);                       // 4096*4096*2  = 33.6 MB
  _Float16* FT    = (_Float16*)(ws + 33554432);            // 2048*4096*2  = 16.8 MB
  float*    field = (float*)   (ws + 50331648);            // 2048*4096*4  = 33.6 MB
  float*    state = (float*)   (ws + 83886080);            // 33.6 MB
  float*    LF    = (float*)   (ws + 117440512);           // 4096*2048*4  = 33.6 MB

  adjacency_kernel<<<N_, 256, 0, stream>>>(node_emb, A);
  encode_kernel<<<(B_ * N_) / 256, 256, 0, stream>>>(
      hist, tid_emb, diw_emb, t2f_w, t2f_b, enc_w1, enc_b1, enc_w2, enc_b2,
      field, FT, state);

  for (int s = 0; s < STEPS_; ++s) {
    gemm_lf_kernel<<<dim3(N_ / 128, COLS_ / 128), 256, 0, stream>>>(A, FT, LF);
    pde_step_kernel<<<(B_ * N_) / 256, 256, 0, stream>>>(
        LF, pde_mix, pde_w1, pde_b1, pde_w2, pde_b2,
        ss_win, ss_wst, ss_b, ss_wout, ss_bout, field, FT, state);
  }

  decode_kernel<<<(B_ * N_) / 256, 256, 0, stream>>>(
      field, dec_w1, dec_b1, dec_w2, dec_b2, (float*)d_out);
}
